// GeneralConv_49615462204048
// MI455X (gfx1250) — matrix operand order repaired and checked by reading, not compiled, not run on hardware
//
#include <hip/hip_runtime.h>

// ---------------------------------------------------------------------------
// Types for CDNA5 WMMA (wave32): v_wmma_f32_16x16x32_bf16
// ---------------------------------------------------------------------------
typedef __attribute__((ext_vector_type(16))) __bf16 v16bf;
typedef __attribute__((ext_vector_type(8)))  float  v8f;

#define BN_EPS 1e-5f
#define NEG_FLT_MAX (-3.402823466e38f)

// ---------------------------------------------------------------------------
// Fragment loaders; this code's contiguous k order (k = 16*(lane/16) + e), same in A and B.
// A (16x32 bf16, MxK): lanes 0-15 row M=lane, elem e = K (k0+e);
//   lanes 16-31 same rows, K offset +16 (matches B).
//   -> two contiguous 16B loads per lane.
// B (32x16 bf16, KxN): lane holds col n = lane&15, elem e = K (kb+e),
//   kb = k0 (+16 for lanes 16-31). With W stored transposed [Nout][K],
//   the 16 elems are contiguous -> two 16B loads.
// C/D (16x16 f32): vgpr r: lanes0-15 -> M=r, N=lane; lanes16-31 -> M=r+8.
// ---------------------------------------------------------------------------
__device__ __forceinline__ v16bf load_a_tile(const __bf16* base, int ld, int lane, int k0) {
    int row  = lane & 15;
    int koff = k0 + ((lane & 16) ? 16 : 0);
    const __bf16* p = base + row * ld + koff;
    union { uint4 u[2]; v16bf v; } cv;
    cv.u[0] = *(const uint4*)(p);
    cv.u[1] = *(const uint4*)(p + 8);
    return cv.v;
}

__device__ __forceinline__ v16bf load_b_tile(const __bf16* Wt, int K, int lane, int k0, int n0) {
    int n  = n0 + (lane & 15);
    int kb = k0 + ((lane & 16) ? 16 : 0);
    const __bf16* p = Wt + (size_t)n * K + kb;
    union { uint4 u[2]; v16bf v; } cv;
    cv.u[0] = *(const uint4*)(p);
    cv.u[1] = *(const uint4*)(p + 8);
    return cv.v;
}

__device__ __forceinline__ void atomicMaxFloat(float* addr, float val) {
    if (val >= 0.0f) atomicMax((int*)addr, __float_as_int(val));
    else             atomicMin((unsigned int*)addr, __float_as_uint(val));
}

// ---------------------------------------------------------------------------
// Generic edge MLP: z = cat([featA[idxA[e]], featB[idxB[e]], vfeat[e]]) (192)
//   y = relu(z @ W1 + b1) @ W2 + b2   (192->128->64), optional relu on z.
// Outputs: bf16 m, and/or f32, and/or atomic segment-max of m*t keyed by dmax.
// 128 threads = 4 waves; each wave owns a 16-edge row tile; 64 WMMAs / wave.
// ---------------------------------------------------------------------------
__global__ __launch_bounds__(128)
void edge_mlp_kernel(const __bf16* __restrict__ featA, const __bf16* __restrict__ featB,
                     const int* __restrict__ idxA, const int* __restrict__ idxB,
                     const __bf16* __restrict__ vfeat,
                     const __bf16* __restrict__ W1t, const float* __restrict__ b1,
                     const __bf16* __restrict__ W2t, const float* __restrict__ b2,
                     int E, int relu_in,
                     __bf16* __restrict__ m_out, float* __restrict__ f_out,
                     float* __restrict__ segmax, const int* __restrict__ dmax_idx,
                     const float* __restrict__ t_ptr)
{
    __shared__ __bf16 zs[4][16][200];   // 16 x 192 input tile per wave (+pad)
    __shared__ __bf16 z2[4][16][136];   // 16 x 128 hidden tile per wave (+pad)

    const int tid  = threadIdx.x;
    const int w    = tid >> 5;
    const int lane = tid & 31;
    const int eb   = blockIdx.x * 64 + w * 16;

    // ---- stage z tile: 16 rows x 24 16B-chunks (dstfeat | srcfeat | edgefeat)
    for (int i = lane; i < 16 * 24; i += 32) {
        int r = i / 24, c = i % 24;
        int e = eb + r; if (e >= E) e = E - 1;
        const __bf16* sp;
        if (c < 8)       sp = featA + (size_t)idxA[e] * 64 + c * 8;
        else if (c < 16) sp = featB + (size_t)idxB[e] * 64 + (c - 8) * 8;
        else             sp = vfeat + (size_t)e * 64 + (c - 16) * 8;
        union { uint4 u; __bf16 b[8]; } cv;
        cv.u = *(const uint4*)sp;
        if (relu_in) {
            #pragma unroll
            for (int j = 0; j < 8; j++) {
                float f = (float)cv.b[j];
                cv.b[j] = (__bf16)fmaxf(f, 0.0f);
            }
        }
        *(uint4*)&zs[w][r][c * 8] = cv.u;
    }
    __syncthreads();

    // ---- layer 1: [16x192] @ [192x128] + b1, relu -> z2 (bf16)
    v16bf a1[6];
    #pragma unroll
    for (int kt = 0; kt < 6; kt++) a1[kt] = load_a_tile(&zs[w][0][0], 200, lane, kt * 32);

    #pragma unroll
    for (int nt = 0; nt < 8; nt++) {
        float bias = b1[nt * 16 + (lane & 15)];
        v8f acc;
        #pragma unroll
        for (int r = 0; r < 8; r++) acc[r] = bias;
        #pragma unroll
        for (int kt = 0; kt < 6; kt++) {
            v16bf bf = load_b_tile(W1t, 192, lane, kt * 32, nt * 16);
            acc = __builtin_amdgcn_wmma_f32_16x16x32_bf16(false, a1[kt], false, bf,
                                                          (short)0, acc, false, false);
        }
        #pragma unroll
        for (int r = 0; r < 8; r++) {
            int row = (lane < 16) ? r : (r + 8);
            z2[w][row][nt * 16 + (lane & 15)] = (__bf16)fmaxf(acc[r], 0.0f);
        }
    }
    __syncthreads();

    // ---- layer 2: [16x128] @ [128x64] + b2
    float tval = t_ptr ? t_ptr[0] : 1.0f;
    v16bf a2[4];
    #pragma unroll
    for (int kt = 0; kt < 4; kt++) a2[kt] = load_a_tile(&z2[w][0][0], 136, lane, kt * 32);

    #pragma unroll
    for (int nt = 0; nt < 4; nt++) {
        float bias = b2[nt * 16 + (lane & 15)];
        v8f acc;
        #pragma unroll
        for (int r = 0; r < 8; r++) acc[r] = bias;
        #pragma unroll
        for (int kt = 0; kt < 4; kt++) {
            v16bf bf = load_b_tile(W2t, 128, lane, kt * 32, nt * 16);
            acc = __builtin_amdgcn_wmma_f32_16x16x32_bf16(false, a2[kt], false, bf,
                                                          (short)0, acc, false, false);
        }
        #pragma unroll
        for (int r = 0; r < 8; r++) {
            int row = (lane < 16) ? r : (r + 8);
            int e = eb + row;
            if (e < E) {
                int col = nt * 16 + (lane & 15);
                float mv = acc[r];
                if (m_out) m_out[(size_t)e * 64 + col] = (__bf16)mv;
                if (f_out) f_out[(size_t)e * 64 + col] = mv;
                if (segmax) atomicMaxFloat(&segmax[(size_t)dmax_idx[e] * 64 + col], mv * tval);
            }
        }
    }
}

// ---------------------------------------------------------------------------
// Node GEMM: y = in[N,128] @ Wt'(=[Nout,128] transposed bf16) + bias.
// Optionally writes y (f32), y (bf16), and accumulates per-feature BN stats
// (sum, sumsq) via shuffle/atomic reduction. 4 waves x 16 rows per block.
// ---------------------------------------------------------------------------
__global__ __launch_bounds__(128)
void node_gemm_kernel(const __bf16* __restrict__ in, int N,
                      const __bf16* __restrict__ Wt, const float* __restrict__ bias, int Nout,
                      float* __restrict__ y_out, __bf16* __restrict__ ybf_out,
                      float* __restrict__ stats)
{
    __shared__ __bf16 zs[4][16][136];   // 16 x 128 per wave

    const int tid  = threadIdx.x;
    const int w    = tid >> 5;
    const int lane = tid & 31;
    const int rb   = blockIdx.x * 64 + w * 16;

    for (int i = lane; i < 16 * 16; i += 32) {
        int r = i / 16, c = i % 16;
        int n = rb + r; if (n >= N) n = N - 1;
        *(uint4*)&zs[w][r][c * 8] = *(const uint4*)(in + (size_t)n * 128 + c * 8);
    }
    __syncthreads();

    v16bf a[4];
    #pragma unroll
    for (int kt = 0; kt < 4; kt++) a[kt] = load_a_tile(&zs[w][0][0], 136, lane, kt * 32);

    for (int nt = 0; nt < Nout / 16; nt++) {
        float bias_v = bias[nt * 16 + (lane & 15)];
        v8f acc;
        #pragma unroll
        for (int r = 0; r < 8; r++) acc[r] = bias_v;
        #pragma unroll
        for (int kt = 0; kt < 4; kt++) {
            v16bf bf = load_b_tile(Wt, 128, lane, kt * 32, nt * 16);
            acc = __builtin_amdgcn_wmma_f32_16x16x32_bf16(false, a[kt], false, bf,
                                                          (short)0, acc, false, false);
        }
        float s1 = 0.0f, s2 = 0.0f;
        #pragma unroll
        for (int r = 0; r < 8; r++) {
            int row = (lane < 16) ? r : (r + 8);
            int n = rb + row;
            if (n < N) {
                int col = nt * 16 + (lane & 15);
                float v = acc[r];
                if (y_out)   y_out[(size_t)n * Nout + col] = v;
                if (ybf_out) ybf_out[(size_t)n * Nout + col] = (__bf16)v;
                s1 += v; s2 += v * v;
            }
        }
        if (stats) {
            s1 += __shfl_down(s1, 16, 32);
            s2 += __shfl_down(s2, 16, 32);
            if (lane < 16) {
                atomicAdd(&stats[nt * 16 + lane], s1);
                atomicAdd(&stats[Nout + nt * 16 + lane], s2);
            }
        }
    }
}

// ---------------------------------------------------------------------------
// Per-column sum / sum-of-squares for BN over the batch dim (C=64).
// ---------------------------------------------------------------------------
__global__ void col_stats_kernel(const float* __restrict__ x, long long rows, int C,
                                 float* __restrict__ stats)
{
    __shared__ float ls[512];
    int tid  = threadIdx.x;          // 256
    int f    = tid & (C - 1);
    int rgrp = tid / C;
    int rpb  = 256 / C;
    float s = 0.0f, s2 = 0.0f;
    for (long long r = (long long)blockIdx.x * rpb + rgrp; r < rows;
         r += (long long)gridDim.x * rpb) {
        float v = x[r * C + f];
        s += v; s2 += v * v;
    }
    ls[tid] = s; ls[256 + tid] = s2;
    __syncthreads();
    if (tid < C) {
        float a1 = ls[tid], a2 = ls[256 + tid];
        for (int j = 1; j < rpb; j++) { a1 += ls[tid + j * C]; a2 += ls[256 + tid + j * C]; }
        atomicAdd(&stats[tid], a1);
        atomicAdd(&stats[C + tid], a2);
    }
}

// BN normalize (+relu) -> bf16.  stats = [sum(C), sumsq(C)].
__global__ void bn_relu_kernel(const float* __restrict__ y, long long rows, int C,
                               const float* __restrict__ stats,
                               const float* __restrict__ g, const float* __restrict__ b,
                               __bf16* __restrict__ out)
{
    long long total = rows * C;
    float inv = 1.0f / (float)rows;
    for (long long i = (long long)blockIdx.x * blockDim.x + threadIdx.x; i < total;
         i += (long long)gridDim.x * blockDim.x) {
        int f = (int)(i % C);
        float mean = stats[f] * inv;
        float var  = stats[C + f] * inv - mean * mean;
        float val  = (y[i] - mean) * rsqrtf(var + BN_EPS) * g[f] + b[f];
        out[i] = (__bf16)fmaxf(val, 0.0f);
    }
}

// Fused scatter-softmax accumulation: denom += exp(m*t - mx), num += m*exp(...)
__global__ void edge_soft_kernel(const __bf16* __restrict__ m, const int* __restrict__ dst,
                                 long long E, const float* __restrict__ segmax,
                                 const float* __restrict__ t_ptr,
                                 float* __restrict__ denom, float* __restrict__ num)
{
    long long total = E * 64;
    float t = t_ptr[0];
    for (long long i = (long long)blockIdx.x * blockDim.x + threadIdx.x; i < total;
         i += (long long)gridDim.x * blockDim.x) {
        long long e = i >> 6; int f = (int)(i & 63);
        float mv = (float)m[i];
        int d = dst[e];
        float ex = __expf(mv * t - segmax[(size_t)d * 64 + f]);
        atomicAdd(&denom[(size_t)d * 64 + f], ex);
        atomicAdd(&num[(size_t)d * 64 + f], mv * ex);
    }
}

// u = cat([h, msg]) with msg = num/denom (0 for isolated nodes), bf16 [N,128]
__global__ void build_u_kernel(const __bf16* __restrict__ h, const float* __restrict__ num,
                               const float* __restrict__ denom, long long N,
                               __bf16* __restrict__ u)
{
    long long total = N * 64;
    for (long long i = (long long)blockIdx.x * blockDim.x + threadIdx.x; i < total;
         i += (long long)gridDim.x * blockDim.x) {
        long long n = i >> 6; int f = (int)(i & 63);
        u[n * 128 + f] = h[i];
        float d = denom[i];
        u[n * 128 + 64 + f] = (__bf16)(d > 0.0f ? num[i] / d : 0.0f);
    }
}

// Convert fp32 weight [K,Nout] to transposed bf16 [Nout,K] (WMMA B fast path)
__global__ void conv_transpose_kernel(const float* __restrict__ wsrc, __bf16* __restrict__ wt,
                                      int K, int Nout)
{
    int total = K * Nout;
    for (int i = blockIdx.x * blockDim.x + threadIdx.x; i < total;
         i += gridDim.x * blockDim.x) {
        int k = i / Nout, n = i % Nout;
        wt[(size_t)n * K + k] = (__bf16)wsrc[i];
    }
}

__global__ void fill_kernel(float* __restrict__ p, float v, long long n)
{
    for (long long i = (long long)blockIdx.x * blockDim.x + threadIdx.x; i < n;
         i += (long long)gridDim.x * blockDim.x) p[i] = v;
}

// ---------------------------------------------------------------------------
extern "C" void kernel_launch(void* const* d_in, const int* in_sizes, int n_in,
                              void* d_out, int out_size, void* d_ws, size_t ws_size,
                              hipStream_t stream)
{
    const float* x    = (const float*)d_in[0];
    const int*   ei   = (const int*)d_in[1];
    const float* ea   = (const float*)d_in[2];
    const float* gh_w = (const float*)d_in[3],  *gh_b = (const float*)d_in[4];
    const float* gv_w = (const float*)d_in[5],  *gv_b = (const float*)d_in[6];
    const float* tpar = (const float*)d_in[7];
    const float* p_w1 = (const float*)d_in[8],  *p_b1 = (const float*)d_in[9];
    const float* p_w2 = (const float*)d_in[10], *p_b2 = (const float*)d_in[11];
    const float* h_w1 = (const float*)d_in[12], *h_b1 = (const float*)d_in[13];
    const float* h_g1 = (const float*)d_in[14], *h_bb1 = (const float*)d_in[15];
    const float* h_w2 = (const float*)d_in[16], *h_b2 = (const float*)d_in[17];
    const float* h_g2 = (const float*)d_in[18], *h_bb2 = (const float*)d_in[19];
    const float* h_w3 = (const float*)d_in[20], *h_b3 = (const float*)d_in[21];
    const float* h_g3 = (const float*)d_in[22], *h_bb3 = (const float*)d_in[23];
    const float* h_w4 = (const float*)d_in[24], *h_b4 = (const float*)d_in[25];
    const float* e_w1 = (const float*)d_in[26], *e_b1 = (const float*)d_in[27];
    const float* e_w2 = (const float*)d_in[28], *e_b2 = (const float*)d_in[29];

    const int N = in_sizes[0] / 64;
    const int E = in_sizes[1] / 2;
    const int* src = ei;
    const int* dst = ei + E;

    // ---- workspace carve-up
    char* ws = (char*)d_ws;
    size_t off = 0;
    auto alloc = [&](size_t bytes) -> void* {
        void* p = ws + off;
        off += (bytes + 255) & ~(size_t)255;
        return p;
    };
    __bf16* h_bf   = (__bf16*)alloc((size_t)N * 64 * 2);
    __bf16* v_bf   = (__bf16*)alloc((size_t)E * 64 * 2);
    __bf16* m_bf   = (__bf16*)alloc((size_t)E * 64 * 2);
    float*  segmax = (float*) alloc((size_t)N * 64 * 4);
    float*  denom  = (float*) alloc((size_t)N * 64 * 4);
    float*  num    = (float*) alloc((size_t)N * 64 * 4);
    __bf16* u_bf   = (__bf16*)alloc((size_t)N * 128 * 2);
    float*  ybuf   = (float*) alloc((size_t)N * 128 * 4);
    __bf16* ho_bf  = (__bf16*)alloc((size_t)N * 64 * 2);
    float*  stats  = (float*) alloc(1024 * 4);
    float* st_x  = stats;        // 128
    float* st_ea = stats + 128;  // 128
    float* st_1  = stats + 256;  // 256
    float* st_2  = stats + 512;  // 256
    float* st_3  = stats + 768;  // 256
    __bf16* wp1t = (__bf16*)alloc((size_t)192 * 128 * 2);
    __bf16* wp2t = (__bf16*)alloc((size_t)128 * 64 * 2);
    __bf16* wh1t = (__bf16*)alloc((size_t)128 * 128 * 2);
    __bf16* wh2t = (__bf16*)alloc((size_t)128 * 128 * 2);
    __bf16* wh3t = (__bf16*)alloc((size_t)128 * 128 * 2);
    __bf16* wh4t = (__bf16*)alloc((size_t)128 * 64 * 2);
    __bf16* we1t = (__bf16*)alloc((size_t)192 * 128 * 2);
    __bf16* we2t = (__bf16*)alloc((size_t)128 * 64 * 2);

    float* h_out = (float*)d_out;
    float* v_out = (float*)d_out + (size_t)N * 64;

    // ---- init scratch
    fill_kernel<<<16, 256, 0, stream>>>(stats, 0.0f, 1024);
    fill_kernel<<<1024, 256, 0, stream>>>(segmax, NEG_FLT_MAX, (long long)N * 64);
    fill_kernel<<<1024, 256, 0, stream>>>(denom, 0.0f, (long long)N * 64);
    fill_kernel<<<1024, 256, 0, stream>>>(num, 0.0f, (long long)N * 64);

    // ---- weights -> transposed bf16
    conv_transpose_kernel<<<96, 256, 0, stream>>>(p_w1, wp1t, 192, 128);
    conv_transpose_kernel<<<32, 256, 0, stream>>>(p_w2, wp2t, 128, 64);
    conv_transpose_kernel<<<64, 256, 0, stream>>>(h_w1, wh1t, 128, 128);
    conv_transpose_kernel<<<64, 256, 0, stream>>>(h_w2, wh2t, 128, 128);
    conv_transpose_kernel<<<64, 256, 0, stream>>>(h_w3, wh3t, 128, 128);
    conv_transpose_kernel<<<32, 256, 0, stream>>>(h_w4, wh4t, 128, 64);
    conv_transpose_kernel<<<96, 256, 0, stream>>>(e_w1, we1t, 192, 128);
    conv_transpose_kernel<<<32, 256, 0, stream>>>(e_w2, we2t, 128, 64);

    // ---- input BN + relu -> bf16
    col_stats_kernel<<<2048, 256, 0, stream>>>(x, N, 64, st_x);
    col_stats_kernel<<<2048, 256, 0, stream>>>(ea, E, 64, st_ea);
    bn_relu_kernel<<<2048, 256, 0, stream>>>(x, N, 64, st_x, gh_w, gh_b, h_bf);
    bn_relu_kernel<<<2048, 256, 0, stream>>>(ea, E, 64, st_ea, gv_w, gv_b, v_bf);

    // ---- message MLP: z = cat([h[dst], h[src], v]) -> m [E,64] + segment max
    edge_mlp_kernel<<<(E + 63) / 64, 128, 0, stream>>>(
        h_bf, h_bf, dst, src, v_bf, wp1t, p_b1, wp2t, p_b2, E, /*relu_in=*/0,
        m_bf, nullptr, segmax, dst, tpar);

    // ---- scatter-softmax fused numerator/denominator
    edge_soft_kernel<<<4096, 256, 0, stream>>>(m_bf, dst, E, segmax, tpar, denom, num);

    // ---- node update MLP (3x GEMM+BN+relu, then final GEMM)
    build_u_kernel<<<2048, 256, 0, stream>>>(h_bf, num, denom, N, u_bf);
    node_gemm_kernel<<<(N + 63) / 64, 128, 0, stream>>>(u_bf, N, wh1t, h_b1, 128, ybuf, nullptr, st_1);
    bn_relu_kernel<<<2048, 256, 0, stream>>>(ybuf, N, 128, st_1, h_g1, h_bb1, u_bf);
    node_gemm_kernel<<<(N + 63) / 64, 128, 0, stream>>>(u_bf, N, wh2t, h_b2, 128, ybuf, nullptr, st_2);
    bn_relu_kernel<<<2048, 256, 0, stream>>>(ybuf, N, 128, st_2, h_g2, h_bb2, u_bf);
    node_gemm_kernel<<<(N + 63) / 64, 128, 0, stream>>>(u_bf, N, wh3t, h_b3, 128, ybuf, nullptr, st_3);
    bn_relu_kernel<<<2048, 256, 0, stream>>>(ybuf, N, 128, st_3, h_g3, h_bb3, u_bf);
    node_gemm_kernel<<<(N + 63) / 64, 128, 0, stream>>>(u_bf, N, wh4t, h_b4, 64, h_out, ho_bf, nullptr);

    // ---- edge update MLP: ev = relu(cat([h_out[src], h_out[dst], v]))
    edge_mlp_kernel<<<(E + 63) / 64, 128, 0, stream>>>(
        ho_bf, ho_bf, src, dst, v_bf, we1t, e_b1, we2t, e_b2, E, /*relu_in=*/1,
        nullptr, v_out, nullptr, nullptr, nullptr);
}
